// AttentionHead_51238959841979
// MI455X (gfx1250) — matrix operand order repaired and checked by reading, not compiled, not run on hardware
//
#include <hip/hip_runtime.h>
#include <hip/hip_bf16.h>
#include <stdint.h>

// Problem constants (match the reference)
#define B_   8
#define S_   2048
#define DIN_ 1024
#define DH_  128    // DOUT / head dim

typedef __attribute__((ext_vector_type(16))) __bf16       v16bf;
typedef __attribute__((ext_vector_type(8)))  float        v8f;
typedef __attribute__((ext_vector_type(4)))  unsigned int u32x4;
typedef __attribute__((ext_vector_type(4)))  float        f32x4;

union Frag32B { v16bf v; u32x4 q[2]; };
union Pack16  { __bf16 h[16]; u32x4 q[2]; };

static __device__ __forceinline__ v8f vzero8() {
  v8f v;
#pragma unroll
  for (int i = 0; i < 8; ++i) v[i] = 0.0f;
  return v;
}

// A-fragment (16x32 bf16), row-major source: lane = (half=lane>>4, row=lane&15).
// elems 0..7  = K[kbase + half*8 .. +8), elems 8..15 = K[kbase + 16 + half*8 .. +8)
static __device__ __forceinline__ v16bf load_a_frag(const __bf16* row_ptr, int kbase, int half) {
  Frag32B f;
  f.q[0] = *reinterpret_cast<const u32x4*>(row_ptr + kbase + half * 8);
  f.q[1] = *reinterpret_cast<const u32x4*>(row_ptr + kbase + 16 + half * 8);
  return f.v;
}

// B-fragment (32x16 bf16) from row-major [N][K] source: lane supplies row N,
// elems 0..7 = K[kbase + half*8 .. +8), elems 8..15 = K[kbase + 16 + half*8 .. +8)
static __device__ __forceinline__ v16bf load_b_frag(const __bf16* row_ptr, int kbase, int half) {
  Frag32B f;
  f.q[0] = *reinterpret_cast<const u32x4*>(row_ptr + kbase + half * 8);
  f.q[1] = *reinterpret_cast<const u32x4*>(row_ptr + kbase + 16 + half * 8);
  return f.v;
}

// ---------------------------------------------------------------------------
// Kernel 1: fused Q/K/V projection.  x[16384,1024]f32 @ W^T[1024,128] + b.
// Q,K stored row-major bf16 [b*S + t][d]; V stored TRANSPOSED bf16 Vt[b][d][t]
// so the attention kernel's PV B-fragments are contiguous global reads.
// 128 blocks x 256 threads; block owns 128 rows; wave w owns rows w*16..w*16+15.
// ---------------------------------------------------------------------------
__global__ __launch_bounds__(256) void qkv_proj_kernel(
    const float* __restrict__ x,
    const float* __restrict__ Wq, const float* __restrict__ bq,
    const float* __restrict__ Wk, const float* __restrict__ bk,
    const float* __restrict__ Wv, const float* __restrict__ bv,
    __bf16* __restrict__ Q, __bf16* __restrict__ K, __bf16* __restrict__ Vt)
{
  __shared__ __bf16 xs[128][32];   // x slice, bf16
  __shared__ __bf16 wsl[128][32];  // W slice, bf16

  const int tid  = threadIdx.x;
  const int lane = tid & 31;
  const int wave = tid >> 5;
  const int half = lane >> 4;
  const int lrow = lane & 15;
  const int mrow0 = blockIdx.x * 128;

  const float* Ws[3] = {Wq, Wk, Wv};
  const float* bs[3] = {bq, bk, bv};

  const int srow = tid >> 1;         // staging row 0..127
  const int scol = (tid & 1) * 16;   // staging col base (16 elems)

  for (int mat = 0; mat < 3; ++mat) {
    v8f acc[8];
#pragma unroll
    for (int j = 0; j < 8; ++j) acc[j] = vzero8();
    const float* W = Ws[mat];

    for (int ks = 0; ks < 32; ++ks) {
      const int kb = ks * 32;
      __syncthreads();  // protect LDS from previous iteration's readers
      {
        const float* xp = x + (size_t)(mrow0 + srow) * DIN_ + kb + scol;
        const float* wp = W + (size_t)srow * DIN_ + kb + scol;
        Pack16 px, pw;
#pragma unroll
        for (int c = 0; c < 4; ++c) {
          f32x4 xv = *reinterpret_cast<const f32x4*>(xp + c * 4);
          f32x4 wv = *reinterpret_cast<const f32x4*>(wp + c * 4);
#pragma unroll
          for (int e = 0; e < 4; ++e) {
            px.h[c * 4 + e] = (__bf16)xv[e];
            pw.h[c * 4 + e] = (__bf16)wv[e];
          }
        }
        *reinterpret_cast<u32x4*>(&xs[srow][scol])      = px.q[0];
        *reinterpret_cast<u32x4*>(&xs[srow][scol + 8])  = px.q[1];
        *reinterpret_cast<u32x4*>(&wsl[srow][scol])     = pw.q[0];
        *reinterpret_cast<u32x4*>(&wsl[srow][scol + 8]) = pw.q[1];
      }
      __syncthreads();

      v16bf aF = load_a_frag(&xs[wave * 16 + lrow][0], 0, half);
#pragma unroll
      for (int j = 0; j < 8; ++j) {
        v16bf bF = load_b_frag(&wsl[j * 16 + lrow][0], 0, half);
        acc[j] = __builtin_amdgcn_wmma_f32_16x16x32_bf16(
            false, aF, false, bF, (short)0, acc[j], false, false);
      }
    }

    // bias add (all 8 acc components of a lane share column N = j*16+lrow) + store
    const float* bias = bs[mat];
    if (mat < 2) {
      // Q / K: row-major [b*S + t][d]
      __bf16* outp = (mat == 0) ? Q : K;
#pragma unroll
      for (int j = 0; j < 8; ++j) {
        const float bval = bias[j * 16 + lrow];
#pragma unroll
        for (int r = 0; r < 8; ++r) {
          const int gr = mrow0 + wave * 16 + r + 8 * half;  // C layout: M = r + 8*half
          outp[(size_t)gr * DH_ + j * 16 + lrow] = (__bf16)(acc[j][r] + bval);
        }
      }
    } else {
      // V: transposed store Vt[b][d][t].  For a fixed lane, components r=0..7 are
      // 8 consecutive rows (t values) -> one contiguous 16B store per column tile.
      const int gr0 = mrow0 + wave * 16 + 8 * half;  // first of 8 consecutive t's
      const int b2  = gr0 >> 11;                     // / S_
      const int t0  = gr0 & (S_ - 1);
#pragma unroll
      for (int j = 0; j < 8; ++j) {
        const float bval = bias[j * 16 + lrow];
        Pack16 pk;
#pragma unroll
        for (int r = 0; r < 8; ++r) pk.h[r] = (__bf16)(acc[j][r] + bval);
        *reinterpret_cast<u32x4*>(
            Vt + ((size_t)b2 * DH_ + j * 16 + lrow) * S_ + t0) = pk.q[0];
      }
    }
  }
}

// ---------------------------------------------------------------------------
// Kernel 2: fused masked attention with online softmax (flash style).
// One block per (batch, 128-query tile); wave w owns query rows w*16..w*16+15.
// No __syncthreads: K and Vt B-fragments come straight from global (L1/L2
// resident), and the P relayout LDS region is private per wave.
// ---------------------------------------------------------------------------
__global__ __launch_bounds__(256) void attn_kernel(
    const __bf16* __restrict__ Q, const __bf16* __restrict__ K,
    const __bf16* __restrict__ Vt, const int* __restrict__ mask,
    float* __restrict__ out)
{
  __shared__ __bf16 ps[8][16][128];  // per-wave P tile, [qrow][key] (32 KB)

  const int tid  = threadIdx.x;
  const int lane = tid & 31;
  const int wave = tid >> 5;
  const int half = lane >> 4;
  const int lrow = lane & 15;

  const int b     = blockIdx.x >> 4;
  const int qt    = blockIdx.x & 15;
  const int qbase = qt * 128;
  const size_t bb  = (size_t)b * S_;          // row base for Q/K
  const size_t vtb = (size_t)b * DH_ * S_;    // base for Vt[b][d][t]

  // Q fragments for this wave's 16 rows, kept in registers for the whole kernel
  v16bf qf[4];
  {
    const __bf16* qrow = Q + (bb + qbase + wave * 16 + lrow) * DH_;
#pragma unroll
    for (int kf = 0; kf < 4; ++kf) qf[kf] = load_a_frag(qrow, kf * 32, half);
  }

  float mrow[8], lsum[8];
#pragma unroll
  for (int r = 0; r < 8; ++r) { mrow[r] = -3.0e38f; lsum[r] = 0.0f; }
  v8f O[8];
#pragma unroll
  for (int d = 0; d < 8; ++d) O[d] = vzero8();

  const float scl = 0.022097086912079608f;  // 1/sqrt(S) — reference's sqrt(seq_len)

  for (int kt = 0; kt < 16; ++kt) {
    const int kb = kt * 128;

    // Prefetch next key tile (K rows and Vt columns) into cache while we
    // compute on this one.  Lowers to global_prefetch_b8 on gfx1250.
    if (kt + 1 < 16) {
      const int kbn = kb + 128;
#pragma unroll
      for (int i = 0; i < 4; ++i) {
        __builtin_prefetch(K + (bb + kbn + lane + 32 * i) * DH_, 0, 3);
        __builtin_prefetch(Vt + vtb + (size_t)(lane + 32 * i) * S_ + kbn, 0, 3);
      }
    }

    // S = Q @ K^T  (K is row-major [t][d] -> already B-fragment friendly)
    v8f sc[8];
#pragma unroll
    for (int nt = 0; nt < 8; ++nt) {
      sc[nt] = vzero8();
      const __bf16* krow = K + (bb + kb + nt * 16 + lrow) * DH_;
#pragma unroll
      for (int kf = 0; kf < 4; ++kf) {
        v16bf bF = load_b_frag(krow, kf * 32, half);
        sc[nt] = __builtin_amdgcn_wmma_f32_16x16x32_bf16(
            false, qf[kf], false, bF, (short)0, sc[nt], false, false);
      }
    }

    // scale + key mask (mask col = N = lane&15 within each 16-wide tile)
#pragma unroll
    for (int nt = 0; nt < 8; ++nt) {
      const int mk = mask[(size_t)b * S_ + kb + nt * 16 + lrow];
#pragma unroll
      for (int r = 0; r < 8; ++r)
        sc[nt][r] = (mk != 0) ? sc[nt][r] * scl : -1.0e9f;
    }

    // online softmax update; row r of a lane's half = query row r + 8*half
    float alpha[8];
#pragma unroll
    for (int r = 0; r < 8; ++r) {
      float mx = sc[0][r];
#pragma unroll
      for (int nt = 1; nt < 8; ++nt) mx = fmaxf(mx, sc[nt][r]);
#pragma unroll
      for (int s = 1; s <= 8; s <<= 1) mx = fmaxf(mx, __shfl_xor(mx, s, 32));
      const float nm = fmaxf(mrow[r], mx);
      alpha[r] = __expf(mrow[r] - nm);
      mrow[r]  = nm;
    }
#pragma unroll
    for (int r = 0; r < 8; ++r) {
      float rs = 0.0f;
#pragma unroll
      for (int nt = 0; nt < 8; ++nt) {
        const float p = __expf(sc[nt][r] - mrow[r]);
        sc[nt][r] = p;
        rs += p;
      }
#pragma unroll
      for (int s = 1; s <= 8; s <<= 1) rs += __shfl_xor(rs, s, 32);
      lsum[r] = lsum[r] * alpha[r] + rs;
#pragma unroll
      for (int d = 0; d < 8; ++d) O[d][r] *= alpha[r];
    }

    // stash P (bf16) in per-wave LDS to convert C-layout -> A-fragment layout
#pragma unroll
    for (int nt = 0; nt < 8; ++nt)
#pragma unroll
      for (int r = 0; r < 8; ++r)
        ps[wave][r + 8 * half][nt * 16 + lrow] = (__bf16)sc[nt][r];

    // O += P @ V   (B frags read directly from transposed Vt in global)
#pragma unroll
    for (int kf = 0; kf < 4; ++kf) {
      v16bf aP = load_a_frag(&ps[wave][lrow][0], kf * 32, half);
#pragma unroll
      for (int d = 0; d < 8; ++d) {
        const __bf16* vrow = Vt + vtb + (size_t)(d * 16 + lrow) * S_ + kb;
        v16bf bV = load_b_frag(vrow, kf * 32, half);
        O[d] = __builtin_amdgcn_wmma_f32_16x16x32_bf16(
            false, aP, false, bV, (short)0, O[d], false, false);
      }
    }
  }

  // normalize by row sum and store f32 context
#pragma unroll
  for (int d = 0; d < 8; ++d) {
#pragma unroll
    for (int r = 0; r < 8; ++r) {
      const size_t row = bb + qbase + wave * 16 + r + 8 * half;
      out[row * DH_ + d * 16 + lrow] = O[d][r] / lsum[r];
    }
  }
}

// ---------------------------------------------------------------------------
// Host launcher. Inputs: x, Wq, bq, Wk, bk, Wv, bv, attention_mask.
// Workspace: Q/K row-major + Vt transposed, bf16, 4 MB each (12 MB total).
// ---------------------------------------------------------------------------
extern "C" void kernel_launch(void* const* d_in, const int* in_sizes, int n_in,
                              void* d_out, int out_size, void* d_ws, size_t ws_size,
                              hipStream_t stream) {
  const float* x    = (const float*)d_in[0];
  const float* Wq   = (const float*)d_in[1];
  const float* bq   = (const float*)d_in[2];
  const float* Wk   = (const float*)d_in[3];
  const float* bk   = (const float*)d_in[4];
  const float* Wv   = (const float*)d_in[5];
  const float* bv   = (const float*)d_in[6];
  const int*   mask = (const int*)d_in[7];

  const size_t nqkv = (size_t)B_ * S_ * DH_;  // 2,097,152 elems
  __bf16* Q  = (__bf16*)d_ws;
  __bf16* K  = Q + nqkv;
  __bf16* Vt = K + nqkv;

  dim3 block(256);
  dim3 grid1((B_ * S_) / 128);   // 128 blocks
  dim3 grid2(B_ * (S_ / 128));   // 128 blocks

  qkv_proj_kernel<<<grid1, block, 0, stream>>>(x, Wq, bq, Wk, bk, Wv, bv, Q, K, Vt);
  attn_kernel<<<grid2, block, 0, stream>>>(Q, K, Vt, mask, (float*)d_out);
}
